// Moondream3SparseMoeBlock_8804682957001
// MI455X (gfx1250) — matrix operand order repaired and checked by reading, not compiled, not run on hardware
//
#include <hip/hip_runtime.h>
#include <stdint.h>

// ---------------- problem constants (must match reference) ----------------
#define NUM_EXPERTS 8
#define TOPK        2
#define HIDDEN      2048
#define INTER       1024
#define T_TOKENS    2048                 // BATCH * SEQ
#define SLOTS       (T_TOKENS * TOPK)    // 4096 (token, k) assignment slots

// ---------------- workspace layout (bytes) ----------------
#define OFF_COUNTS  0
#define OFF_LIST    256
#define OFF_WTS     (OFF_LIST + NUM_EXPERTS * T_TOKENS * 4)          // 65792
#define OFF_XB      (OFF_WTS + SLOTS * 4)                            // 82176
#define OFF_H       (OFF_XB + (size_t)T_TOKENS * HIDDEN * 2)         // + 8 MiB
#define OFF_WUPT    (OFF_H + (size_t)SLOTS * INTER * 2)              // + 8 MiB
#define OFF_WGT     (OFF_WUPT + (size_t)NUM_EXPERTS * HIDDEN * INTER * 2)  // +32 MiB
#define OFF_WDT     (OFF_WGT + (size_t)NUM_EXPERTS * HIDDEN * INTER * 2)   // +32 MiB
#define OFF_CONTRIB (OFF_WDT + (size_t)NUM_EXPERTS * INTER * HIDDEN * 2)   // +32 MiB
// total ~ 145 MiB

// ---------------- WMMA types ----------------
typedef __attribute__((ext_vector_type(16))) __bf16 v16bf;
typedef __attribute__((ext_vector_type(8)))  float  v8f;

union Frag {
    v16bf          v;
    unsigned short us[16];
    uint4          q[2];
};

// fp32 -> bf16 bits, round-to-nearest-even
__device__ __forceinline__ unsigned short f2bf(float f) {
    unsigned u = __float_as_uint(f);
    u += 0x7FFFu + ((u >> 16) & 1u);
    return (unsigned short)(u >> 16);
}

__device__ __forceinline__ float gelu_exact(float v) {
    return 0.5f * v * (1.0f + erff(v * 0.70710678118654752f));
}

// ---------------- kernel 0: zero per-expert counters ----------------
__global__ void moe_init_kernel(int* __restrict__ counts) {
    if (threadIdx.x < NUM_EXPERTS) counts[threadIdx.x] = 0;
}

// ---------------- kernel 1: router (one wave32 per token) ----------------
__global__ __launch_bounds__(256)
void moe_router_kernel(const float* __restrict__ x,
                       const float* __restrict__ gate_w,
                       const float* __restrict__ gate_b,
                       float* __restrict__ logits_out,   // [T, E]
                       int*   __restrict__ counts,       // [E]
                       int*   __restrict__ list,         // [E, T] slot ids
                       float* __restrict__ wts)          // [SLOTS]
{
    const int wave = threadIdx.x >> 5;
    const int lane = threadIdx.x & 31;
    const int t    = blockIdx.x * 8 + wave;

    float acc[NUM_EXPERTS];
#pragma unroll
    for (int e = 0; e < NUM_EXPERTS; ++e) acc[e] = 0.0f;

    const float* xr = x + (size_t)t * HIDDEN;
    for (int h = lane; h < HIDDEN; h += 32) {
        const float xv = xr[h];
        const float4* g4 = (const float4*)(gate_w + (size_t)h * NUM_EXPERTS);
        const float4 a = g4[0];
        const float4 b = g4[1];
        acc[0] += xv * a.x; acc[1] += xv * a.y; acc[2] += xv * a.z; acc[3] += xv * a.w;
        acc[4] += xv * b.x; acc[5] += xv * b.y; acc[6] += xv * b.z; acc[7] += xv * b.w;
    }
#pragma unroll
    for (int e = 0; e < NUM_EXPERTS; ++e)
#pragma unroll
        for (int off = 16; off > 0; off >>= 1)
            acc[e] += __shfl_xor(acc[e], off, 32);

    if (lane == 0) {
        float lg[NUM_EXPERTS];
#pragma unroll
        for (int e = 0; e < NUM_EXPERTS; ++e) {
            lg[e] = acc[e] + gate_b[e];
            logits_out[(size_t)t * NUM_EXPERTS + e] = lg[e];
        }
        int i0 = 0;
#pragma unroll
        for (int e = 1; e < NUM_EXPERTS; ++e) if (lg[e] > lg[i0]) i0 = e;
        int i1 = (i0 == 0) ? 1 : 0;
#pragma unroll
        for (int e = 0; e < NUM_EXPERTS; ++e)
            if (e != i0 && lg[e] > lg[i1]) i1 = e;

        const float e1  = __expf(lg[i1] - lg[i0]);
        const float inv = 1.0f / (1.0f + e1);
        wts[2 * t + 0] = inv;
        wts[2 * t + 1] = e1 * inv;

        int p0 = atomicAdd(&counts[i0], 1);
        list[i0 * T_TOKENS + p0] = 2 * t + 0;
        int p1 = atomicAdd(&counts[i1], 1);
        list[i1 * T_TOKENS + p1] = 2 * t + 1;
    }
}

// ---------------- kernel 2: x -> bf16 (elementwise, coalesced) ----------------
__global__ __launch_bounds__(256)
void moe_convert_x_kernel(const float* __restrict__ x, unsigned short* __restrict__ xb) {
    const int i = blockIdx.x * 256 + threadIdx.x;     // over T*HIDDEN/4
    const float4 v = ((const float4*)x)[i];
    ushort4 o;
    o.x = f2bf(v.x); o.y = f2bf(v.y); o.z = f2bf(v.z); o.w = f2bf(v.w);
    ((ushort4*)xb)[i] = o;
}

// ---------------- kernel 3: weight convert + transpose via LDS ----------------
// in:  [E][K][N] fp32 (row-major over N)   out: [E][N][K] bf16
__global__ __launch_bounds__(256)
void moe_transpose_bf16_kernel(const float* __restrict__ in,
                               unsigned short* __restrict__ out,
                               int K, int N) {
    __shared__ unsigned short tile[64][66];   // +2 pad against bank conflicts
    const int e  = blockIdx.z;
    const int kb = blockIdx.x * 64;
    const int nb = blockIdx.y * 64;
    const float*    src = in  + (size_t)e * K * N;
    unsigned short* dst = out + (size_t)e * N * K;

    const int t    = threadIdx.x;
    const int col4 = (t & 15) * 4;
    const int r0   = t >> 4;                  // 0..15
#pragma unroll
    for (int p = 0; p < 4; ++p) {
        const int r = r0 + p * 16;
        const float4 v = *(const float4*)(src + (size_t)(kb + r) * N + nb + col4);
        tile[r][col4 + 0] = f2bf(v.x);
        tile[r][col4 + 1] = f2bf(v.y);
        tile[r][col4 + 2] = f2bf(v.z);
        tile[r][col4 + 3] = f2bf(v.w);
    }
    __syncthreads();

    const int n    = t >> 2;                  // 0..63
    const int kseg = (t & 3) * 16;
    union { uint4 q[2]; unsigned short us[16]; } o;
#pragma unroll
    for (int j = 0; j < 16; ++j) o.us[j] = tile[kseg + j][n];
    uint4* dq = (uint4*)(dst + (size_t)(nb + n) * K + kb + kseg);
    dq[0] = o.q[0];
    dq[1] = o.q[1];
}

// ---------------- kernel 4: up/gate GEMM + activation (wave per 16x32 tile) ----------------
// All operands bf16, all fragment loads are aligned b128; 4 WMMAs per K-step.
__global__ __launch_bounds__(32)
void moe_upgate_kernel(const unsigned short* __restrict__ xb,     // [T, HIDDEN] bf16
                       const unsigned short* __restrict__ wupT,   // [E, INTER, HIDDEN] bf16
                       const unsigned short* __restrict__ wgT,    // [E, INTER, HIDDEN] bf16
                       const int*   __restrict__ counts,
                       const int*   __restrict__ list,
                       unsigned short* __restrict__ h_buf)        // [SLOTS, INTER] bf16
{
    const int e     = blockIdx.z;
    const int cnt   = counts[e];
    const int mbase = blockIdx.x * 16;
    if (mbase >= cnt) return;                 // uniform: EXEC stays full for WMMA

    const int lane  = threadIdx.x;
    const int half  = lane >> 4;
    const int lm    = lane & 15;
    const int nbase = blockIdx.y * 32;

    int rowIdx = mbase + lm;
    if (rowIdx >= cnt) rowIdx = cnt - 1;
    const int slot = list[e * T_TOKENS + rowIdx];
    const int tok  = slot >> 1;

    const unsigned short* ab  = xb   + (size_t)tok * HIDDEN;
    const unsigned short* bu0 = wupT + ((size_t)e * INTER + nbase + lm)      * HIDDEN;
    const unsigned short* bu1 = bu0  + (size_t)16 * HIDDEN;
    const unsigned short* bg0 = wgT  + ((size_t)e * INTER + nbase + lm)      * HIDDEN;
    const unsigned short* bg1 = bg0  + (size_t)16 * HIDDEN;

    v8f aU0 = {}, aU1 = {}, aG0 = {}, aG1 = {};

    for (int kb = 0; kb < HIDDEN; kb += 32) {
        Frag A, BU0, BU1, BG0, BG1;
        A.q[0] = *(const uint4*)(ab + kb + half * 16);
        A.q[1] = *(const uint4*)(ab + kb + half * 16 + 8);
        const int ko = kb + half * 16;
        BU0.q[0] = *(const uint4*)(bu0 + ko);
        BU0.q[1] = *(const uint4*)(bu0 + ko + 8);
        BU1.q[0] = *(const uint4*)(bu1 + ko);
        BU1.q[1] = *(const uint4*)(bu1 + ko + 8);
        BG0.q[0] = *(const uint4*)(bg0 + ko);
        BG0.q[1] = *(const uint4*)(bg0 + ko + 8);
        BG1.q[0] = *(const uint4*)(bg1 + ko);
        BG1.q[1] = *(const uint4*)(bg1 + ko + 8);
        if (kb + 128 < HIDDEN) {
            __builtin_prefetch(bu0 + kb + 128, 0, 1);
            __builtin_prefetch(bg0 + kb + 128, 0, 1);
        }
        aU0 = __builtin_amdgcn_wmma_f32_16x16x32_bf16(false, A.v, false, BU0.v, (short)0, aU0, false, false);
        aU1 = __builtin_amdgcn_wmma_f32_16x16x32_bf16(false, A.v, false, BU1.v, (short)0, aU1, false, false);
        aG0 = __builtin_amdgcn_wmma_f32_16x16x32_bf16(false, A.v, false, BG0.v, (short)0, aG0, false, false);
        aG1 = __builtin_amdgcn_wmma_f32_16x16x32_bf16(false, A.v, false, BG1.v, (short)0, aG1, false, false);
    }

#pragma unroll
    for (int r = 0; r < 8; ++r) {
        const int m = r + half * 8;
        const int slotM = __shfl(slot, m, 32);
        if (mbase + m < cnt) {
            const float h0 = gelu_exact(aU0[r]) * (aG0[r] + 1.0f);
            const float h1 = gelu_exact(aU1[r]) * (aG1[r] + 1.0f);
            unsigned short* hr = h_buf + (size_t)slotM * INTER + nbase + lm;
            hr[0]  = f2bf(h0);
            hr[16] = f2bf(h1);
        }
    }
}

// ---------------- kernel 5: down GEMM (wave per 16x64 tile) ----------------
__global__ __launch_bounds__(32)
void moe_down_kernel(const unsigned short* __restrict__ h_buf,  // [SLOTS, INTER] bf16
                     const unsigned short* __restrict__ wdT,    // [E, HIDDEN, INTER] bf16
                     const int*   __restrict__ counts,
                     const int*   __restrict__ list,
                     float* __restrict__ contrib)               // [SLOTS, HIDDEN]
{
    const int e     = blockIdx.z;
    const int cnt   = counts[e];
    const int mbase = blockIdx.x * 16;
    if (mbase >= cnt) return;

    const int lane  = threadIdx.x;
    const int half  = lane >> 4;
    const int lm    = lane & 15;
    const int nbase = blockIdx.y * 64;

    int rowIdx = mbase + lm;
    if (rowIdx >= cnt) rowIdx = cnt - 1;
    const int slot = list[e * T_TOKENS + rowIdx];

    const unsigned short* ar = h_buf + (size_t)slot * INTER;
    const unsigned short* bd0 = wdT + ((size_t)e * HIDDEN + nbase + lm) * INTER;
    const unsigned short* bd1 = bd0 + (size_t)16 * INTER;
    const unsigned short* bd2 = bd0 + (size_t)32 * INTER;
    const unsigned short* bd3 = bd0 + (size_t)48 * INTER;

    v8f a0 = {}, a1 = {}, a2 = {}, a3 = {};

    for (int kb = 0; kb < INTER; kb += 32) {
        Frag A, B0, B1, B2, B3;
        A.q[0] = *(const uint4*)(ar + kb + half * 16);
        A.q[1] = *(const uint4*)(ar + kb + half * 16 + 8);
        const int ko = kb + half * 16;
        B0.q[0] = *(const uint4*)(bd0 + ko);
        B0.q[1] = *(const uint4*)(bd0 + ko + 8);
        B1.q[0] = *(const uint4*)(bd1 + ko);
        B1.q[1] = *(const uint4*)(bd1 + ko + 8);
        B2.q[0] = *(const uint4*)(bd2 + ko);
        B2.q[1] = *(const uint4*)(bd2 + ko + 8);
        B3.q[0] = *(const uint4*)(bd3 + ko);
        B3.q[1] = *(const uint4*)(bd3 + ko + 8);
        if (kb + 128 < INTER)
            __builtin_prefetch(bd0 + kb + 128, 0, 1);
        a0 = __builtin_amdgcn_wmma_f32_16x16x32_bf16(false, A.v, false, B0.v, (short)0, a0, false, false);
        a1 = __builtin_amdgcn_wmma_f32_16x16x32_bf16(false, A.v, false, B1.v, (short)0, a1, false, false);
        a2 = __builtin_amdgcn_wmma_f32_16x16x32_bf16(false, A.v, false, B2.v, (short)0, a2, false, false);
        a3 = __builtin_amdgcn_wmma_f32_16x16x32_bf16(false, A.v, false, B3.v, (short)0, a3, false, false);
    }

#pragma unroll
    for (int r = 0; r < 8; ++r) {
        const int m = r + half * 8;
        const int slotM = __shfl(slot, m, 32);
        if (mbase + m < cnt) {
            float* cr = contrib + (size_t)slotM * HIDDEN + nbase + lm;
            cr[0]  = a0[r];
            cr[16] = a1[r];
            cr[32] = a2[r];
            cr[48] = a3[r];
        }
    }
}

// ---------------- kernel 6: weighted combine (fixed order -> deterministic) ----------------
__global__ __launch_bounds__(256)
void moe_combine_kernel(const float* __restrict__ contrib,
                        const float* __restrict__ wts,
                        float* __restrict__ out)
{
    const int i  = blockIdx.x * blockDim.x + threadIdx.x;  // over T*HIDDEN/4 float4s
    const int t  = i >> 9;                                 // / (HIDDEN/4 = 512)
    const int c4 = i & 511;
    const float w0 = wts[2 * t + 0];
    const float w1 = wts[2 * t + 1];
    const float4 a = ((const float4*)(contrib + (size_t)(2 * t + 0) * HIDDEN))[c4];
    const float4 b = ((const float4*)(contrib + (size_t)(2 * t + 1) * HIDDEN))[c4];
    float4 o;
    o.x = w0 * a.x + w1 * b.x;
    o.y = w0 * a.y + w1 * b.y;
    o.z = w0 * a.z + w1 * b.z;
    o.w = w0 * a.w + w1 * b.w;
    ((float4*)out)[i] = o;
}

// ---------------- host launcher ----------------
extern "C" void kernel_launch(void* const* d_in, const int* in_sizes, int n_in,
                              void* d_out, int out_size, void* d_ws, size_t ws_size,
                              hipStream_t stream) {
    (void)in_sizes; (void)n_in; (void)out_size; (void)ws_size;

    const float* x      = (const float*)d_in[0];  // [B,S,HIDDEN]
    const float* gate_w = (const float*)d_in[1];  // [HIDDEN, E]
    const float* gate_b = (const float*)d_in[2];  // [E]
    const float* w_up   = (const float*)d_in[3];  // [E, HIDDEN, INTER]
    const float* w_gate = (const float*)d_in[4];  // [E, HIDDEN, INTER]
    const float* w_down = (const float*)d_in[5];  // [E, INTER, HIDDEN]

    float* out    = (float*)d_out;                       // [T, HIDDEN]
    float* logits = out + (size_t)T_TOKENS * HIDDEN;     // [T, E]

    char* ws = (char*)d_ws;
    int*            counts  = (int*)(ws + OFF_COUNTS);
    int*            list    = (int*)(ws + OFF_LIST);
    float*          wts     = (float*)(ws + OFF_WTS);
    unsigned short* xb      = (unsigned short*)(ws + OFF_XB);
    unsigned short* h_buf   = (unsigned short*)(ws + OFF_H);
    unsigned short* wupT    = (unsigned short*)(ws + OFF_WUPT);
    unsigned short* wgT     = (unsigned short*)(ws + OFF_WGT);
    unsigned short* wdT     = (unsigned short*)(ws + OFF_WDT);
    float*          contrib = (float*)(ws + OFF_CONTRIB);

    moe_init_kernel<<<1, 32, 0, stream>>>(counts);

    moe_router_kernel<<<T_TOKENS / 8, 256, 0, stream>>>(
        x, gate_w, gate_b, logits, counts, list, wts);

    moe_convert_x_kernel<<<(T_TOKENS * HIDDEN / 4) / 256, 256, 0, stream>>>(x, xb);

    // [E][K][N] fp32 -> [E][N][K] bf16
    moe_transpose_bf16_kernel<<<dim3(HIDDEN / 64, INTER / 64, NUM_EXPERTS), 256, 0, stream>>>(
        w_up, wupT, HIDDEN, INTER);
    moe_transpose_bf16_kernel<<<dim3(HIDDEN / 64, INTER / 64, NUM_EXPERTS), 256, 0, stream>>>(
        w_gate, wgT, HIDDEN, INTER);
    moe_transpose_bf16_kernel<<<dim3(INTER / 64, HIDDEN / 64, NUM_EXPERTS), 256, 0, stream>>>(
        w_down, wdT, INTER, HIDDEN);

    moe_upgate_kernel<<<dim3(T_TOKENS / 16, INTER / 32, NUM_EXPERTS), 32, 0, stream>>>(
        xb, wupT, wgT, counts, list, h_buf);

    moe_down_kernel<<<dim3(T_TOKENS / 16, HIDDEN / 64, NUM_EXPERTS), 32, 0, stream>>>(
        h_buf, wdT, counts, list, contrib);

    moe_combine_kernel<<<(T_TOKENS * HIDDEN / 4) / 256, 256, 0, stream>>>(
        contrib, wts, out);
}
